// RNN_56538949484891
// MI455X (gfx1250) — matrix operand order repaired and checked by reading, not compiled, not run on hardware
//
#include <hip/hip_runtime.h>
#include <hip/hip_bf16.h>
#include <math.h>

// RNN: B=32, T=4096, D=H=256
//  phase0: convert 3 weight mats fp32 [n][k] -> bf16 transposed [k][n] in d_ws
//  phase1: d_out <- Xproj = X @ Wx^T + bx        (WMMA bf16 GEMM, M=131072)
//  phase2: in-place scan over d_out: row-block t: read xp, h=tanh(Wh·h+bh+xp), write h
//          (single workgroup, 16 waves, Wh B-frags resident in VGPRs, h double-buffered in LDS,
//           xp loads hoisted above the WMMA chain, LDS-only split barrier per step)
//  phase3: d_out <- d_out @ Wy^T + by  in place  (WMMA bf16 GEMM)

typedef __attribute__((ext_vector_type(8)))  __bf16 v8bf;
typedef __attribute__((ext_vector_type(16))) __bf16 v16bf;
typedef __attribute__((ext_vector_type(8)))  float  v8f;
typedef __attribute__((ext_vector_type(4)))  float  v4f;

#define T_SEQ 4096
#define NB    32
#define DIM   256
#define LDH   272   // LDS pitch (bf16 elems) for h rows: 544B -> keeps lane banks spread

__device__ __forceinline__ v16bf cat16(v8bf a, v8bf b) {
  v16bf r;
#pragma unroll
  for (int i = 0; i < 8; i++) { r[i] = a[i]; r[i + 8] = b[i]; }
  return r;
}

__device__ __forceinline__ v8bf cvt8(v4f a, v4f b) {
  v8bf r;
#pragma unroll
  for (int i = 0; i < 4; i++) { r[i] = (__bf16)a[i]; r[i + 4] = (__bf16)b[i]; }
  return r;
}

// A-frag (16x32 bf16) from an fp32 row (stride DIM). Per ISA 16-bit A layout:
// lanes 0-15 row=lane K in {kb..kb+7, kb+16..kb+23}; lanes 16-31 same rows, K+8.
__device__ __forceinline__ v16bf load_a_f32(const float* __restrict__ row, int kb, int hs) {
  const v4f* p0 = (const v4f*)(row + kb + 8 * hs);
  const v4f* p1 = (const v4f*)(row + kb + 16 + 8 * hs);
  return cat16(cvt8(p0[0], p0[1]), cvt8(p1[0], p1[1]));
}

// B-frag (32x16 bf16) from K-major bf16 weights Wt[k*DIM + n]: N = lane%16, K in {kb+8h..+7, kb+16+8h..+7}.
__device__ __forceinline__ v16bf load_b(const __bf16* __restrict__ Wt, int kb, int nt, int lane) {
  const int col = nt * 16 + (lane & 15), h = lane >> 4;
  v16bf r;
#pragma unroll
  for (int s = 0; s < 8; s++) {
    r[s] = Wt[(size_t)(kb + 8 * h + s) * DIM + col];
    r[s + 8] = Wt[(size_t)(kb + 16 + 8 * h + s) * DIM + col];
  }
  return r;
}

#define WMMA_BF16(A, Bm, C) \
  __builtin_amdgcn_wmma_f32_16x16x32_bf16(false, (A), false, (Bm), (short)0, (C), false, false)

// LDS-only split barrier: wait DS counter, then workgroup barrier (-1).
// Avoids __syncthreads()'s wait on global STOREcnt (scan's global h-stores are
// never consumed inside the scan, so waiting on them each step is pure latency).
#define LDS_BARRIER() \
  asm volatile("s_wait_dscnt 0x0\n\ts_barrier_signal -1\n\ts_barrier_wait -1" ::: "memory")

// ---------------- phase 0: weight convert + transpose ----------------
__global__ void wconv_kernel(const float* __restrict__ W, __bf16* __restrict__ Wt) {
  int idx = blockIdx.x * blockDim.x + threadIdx.x;   // 65536 threads
  int k = idx >> 8, n = idx & 255;
  Wt[idx] = (__bf16)W[(size_t)n * DIM + k];          // Wt[k][n] = W[n][k]
}

// ---------------- phase 1: Xproj -> d_out ----------------
__global__ __launch_bounds__(256)
void xproj_kernel(const float* __restrict__ X, const __bf16* __restrict__ Wxt,
                  const float* __restrict__ bx, float* __restrict__ out) {
  const int lane = threadIdx.x & 31;
  const int wave = threadIdx.x >> 5;               // 8 waves -> 128 rows / block
  const int lo = lane & 15, hs = lane >> 4;
  const int mbase = blockIdx.x * 128 + wave * 16;
  const int m = mbase + lo;                        // output row = t*32 + b
  const int t = m >> 5, b = m & 31;
  const float* xrow = X + ((size_t)b * T_SEQ + t) * DIM;

  v16bf af[8];
#pragma unroll
  for (int ks = 0; ks < 8; ks++) af[ks] = load_a_f32(xrow, ks * 32, hs);

#pragma unroll
  for (int nt = 0; nt < 16; nt++) {
    const int n = nt * 16 + lo;
    const float bias = bx[n];
    v8f c;
#pragma unroll
    for (int r = 0; r < 8; r++) c[r] = bias;
#pragma unroll
    for (int ks = 0; ks < 8; ks++)
      c = WMMA_BF16(af[ks], load_b(Wxt, ks * 32, nt, lane), c);
    const size_t obase = (size_t)(mbase + 8 * hs) * DIM + n;
#pragma unroll
    for (int r = 0; r < 8; r++) out[obase + (size_t)r * DIM] = c[r];
  }
}

// ---------------- phase 2: sequential scan, in place over d_out ----------------
__global__ __launch_bounds__(512)
void scan_kernel(const __bf16* __restrict__ Wht, const float* __restrict__ bh,
                 float* __restrict__ out) {
  __shared__ __bf16 hA[NB * LDH];                  // double-buffered h (bf16)
  __shared__ __bf16 hB[NB * LDH];
  const int tid = threadIdx.x;
  const int lane = tid & 31;
  const int nt = tid >> 5;                         // 16 waves; wave = n-tile, handles both m-tiles
  const int lo = lane & 15, hs = lane >> 4;
  const int n = nt * 16 + lo;

  for (int i = tid; i < NB * LDH; i += 512) {      // h0 = 0
    hA[i] = (__bf16)0.f; hB[i] = (__bf16)0.f;
  }
  const float bias = bh[n];
  v16bf bw[8];                                     // Wh B-frags live in VGPRs all 4096 steps
#pragma unroll
  for (int ks = 0; ks < 8; ks++) bw[ks] = load_b(Wht, ks * 32, nt, lane);
  __syncthreads();

  auto step = [&](const __bf16* __restrict__ hr, __bf16* __restrict__ hw, int t) {
    const size_t row0 = (size_t)t * NB + 8 * hs;   // out rows t*32 + m

    // Issue all xp loads FIRST: independent of h, latency overlaps the WMMA chain.
    float xp0[8], xp1[8];
#pragma unroll
    for (int r = 0; r < 8; r++) {
      xp0[r] = out[(row0 + r) * DIM + n];
      xp1[r] = out[(row0 + 16 + r) * DIM + n];
    }

    v8f c0, c1;
#pragma unroll
    for (int r = 0; r < 8; r++) { c0[r] = bias; c1[r] = bias; }
#pragma unroll
    for (int ks = 0; ks < 8; ks++) {
      const int kb = ks * 32;
      const v8bf* a0p = (const v8bf*)(hr + (size_t)lo * LDH + kb + 8 * hs);
      const v8bf* a0q = (const v8bf*)(hr + (size_t)lo * LDH + kb + 16 + 8 * hs);
      const v8bf* a1p = (const v8bf*)(hr + (size_t)(16 + lo) * LDH + kb + 8 * hs);
      const v8bf* a1q = (const v8bf*)(hr + (size_t)(16 + lo) * LDH + kb + 16 + 8 * hs);
      c0 = WMMA_BF16(cat16(a0p[0], a0q[0]), bw[ks], c0);
      c1 = WMMA_BF16(cat16(a1p[0], a1q[0]), bw[ks], c1);
    }

#pragma unroll
    for (int r = 0; r < 8; r++) {
      float v = tanhf(c0[r] + xp0[r]);
      out[(row0 + r) * DIM + n] = v;               // fire-and-forget global store
      hw[(size_t)(8 * hs + r) * LDH + n] = (__bf16)v;
      float w = tanhf(c1[r] + xp1[r]);
      out[(row0 + 16 + r) * DIM + n] = w;
      hw[(size_t)(16 + 8 * hs + r) * LDH + n] = (__bf16)w;
    }
    LDS_BARRIER();                                 // only DS counter + barrier on critical path
  };

  for (int t = 0; t < T_SEQ; t += 2) {             // unroll x2: static buffer swap
    step(hA, hB, t);
    step(hB, hA, t + 1);
  }
}

// ---------------- phase 3: Y = H @ Wy^T + by, in place over d_out ----------------
__global__ __launch_bounds__(256)
void yout_kernel(const __bf16* __restrict__ Wyt, const float* __restrict__ by,
                 float* __restrict__ out) {
  const int lane = threadIdx.x & 31;
  const int wave = threadIdx.x >> 5;
  const int lo = lane & 15, hs = lane >> 4;
  const int mbase = blockIdx.x * 128 + wave * 16;
  const float* hrow = out + (size_t)(mbase + lo) * DIM;

  v16bf af[8];                                     // load ALL of h-rows before any store
#pragma unroll
  for (int ks = 0; ks < 8; ks++) af[ks] = load_a_f32(hrow, ks * 32, hs);

#pragma unroll
  for (int nt = 0; nt < 16; nt++) {
    const int n = nt * 16 + lo;
    const float bias = by[n];
    v8f c;
#pragma unroll
    for (int r = 0; r < 8; r++) c[r] = bias;
#pragma unroll
    for (int ks = 0; ks < 8; ks++)
      c = WMMA_BF16(af[ks], load_b(Wyt, ks * 32, nt, lane), c);
    const size_t obase = (size_t)(mbase + 8 * hs) * DIM + n;
#pragma unroll
    for (int r = 0; r < 8; r++) out[obase + (size_t)r * DIM] = c[r];
  }
}

extern "C" void kernel_launch(void* const* d_in, const int* in_sizes, int n_in,
                              void* d_out, int out_size, void* d_ws, size_t ws_size,
                              hipStream_t stream) {
  (void)in_sizes; (void)n_in; (void)out_size; (void)ws_size;
  const float* X   = (const float*)d_in[0];
  const float* Wxw = (const float*)d_in[1];
  const float* Wxb = (const float*)d_in[2];
  const float* Whw = (const float*)d_in[3];
  const float* Whb = (const float*)d_in[4];
  const float* Wyw = (const float*)d_in[5];
  const float* Wyb = (const float*)d_in[6];
  float* out = (float*)d_out;

  __bf16* Wxt = (__bf16*)d_ws;          // 3 x 128KB bf16 weights in workspace
  __bf16* Wht = Wxt + DIM * DIM;
  __bf16* Wyt = Wht + DIM * DIM;

  wconv_kernel<<<256, 256, 0, stream>>>(Wxw, Wxt);
  wconv_kernel<<<256, 256, 0, stream>>>(Whw, Wht);
  wconv_kernel<<<256, 256, 0, stream>>>(Wyw, Wyt);
  xproj_kernel<<<1024, 256, 0, stream>>>(X, Wxt, Wxb, out);   // 131072 rows / 128 per block
  scan_kernel<<<1, 512, 0, stream>>>(Wht, Whb, out);          // one resident workgroup
  yout_kernel<<<1024, 256, 0, stream>>>(Wyt, Wyb, out);
}
